// GRANCascadingPredictor_28252294873251
// MI455X (gfx1250) — matrix operand order repaired and checked by reading, not compiled, not run on hardware
//
#include <hip/hip_runtime.h>
#include <hip/hip_bf16.h>

typedef __bf16 bf16;
typedef __attribute__((ext_vector_type(16))) __bf16 v16bf;
typedef __attribute__((ext_vector_type(8)))  float  v8f;

#define LDS_STR 40            // 32 + 8 pad bf16 elems -> 80B row stride (16B aligned)
#define LDS_TILE (128 * LDS_STR)

__device__ __forceinline__ float sigmoidf_(float x) { return 1.0f / (1.0f + __expf(-x)); }

// ---------------------------------------------------------------------------
// Tiled bf16 WMMA GEMM:  C[M,N] = act(A[M,K] @ B[K,N] + bias)
// A row-major [M,K]; Bt is B TRANSPOSED, row-major [N,K]. K % 32 == 0.
// M assumed multiple of 128 (true for all calls). N arbitrary: out-of-range B
// rows are CLAMPED (their garbage lands only in output columns >= N, which the
// epilogue guard discards).
// Block = 256 threads (8 waves); tile 128x128; each wave computes 32x64
// (2x4 WMMA tiles => 8 v_wmma per BK=32 step). Double-buffered LDS, one
// barrier per BK step; next tile's global loads overlap the WMMA chain.
// ---------------------------------------------------------------------------
__global__ __launch_bounds__(256)
void gran_gemm_bf16(const bf16* __restrict__ A, const bf16* __restrict__ Bt,
                    const float* __restrict__ bias,
                    float* __restrict__ Cf, bf16* __restrict__ Cb,
                    int M, int N, int K, int act)
{
    __shared__ bf16 As[2 * LDS_TILE];
    __shared__ bf16 Bs[2 * LDS_TILE];

    const int t    = threadIdx.x;
    const int lane = t & 31;
    const int wave = t >> 5;
    const int bm   = blockIdx.y * 128;
    const int bn   = blockIdx.x * 128;
    const int wm0  = (wave >> 1) * 32;   // 4 wave-rows
    const int wn0  = (wave & 1) * 64;    // 2 wave-cols
    const int h    = lane >> 4;
    const int l16  = lane & 15;

    v8f acc[2][4];
#pragma unroll
    for (int mi = 0; mi < 2; ++mi)
#pragma unroll
        for (int ni = 0; ni < 4; ++ni)
#pragma unroll
            for (int e = 0; e < 8; ++e) acc[mi][ni][e] = 0.0f;

    // staging: each thread moves 16 bf16 (2 x b128) of A and of Bt per BK step
    const int srow = t >> 1;             // 0..127
    const int koff = (t & 1) << 4;       // 0 or 16
    const int brow = (bn + srow < N) ? (bn + srow) : (N - 1);   // clamp, no guard
    const bf16* Ap = A + (size_t)(bm + srow) * K + koff;
    const bf16* Bp = Bt + (size_t)brow * K + koff;
    bf16* AsW = &As[srow * LDS_STR + koff];
    bf16* BsW = &Bs[srow * LDS_STR + koff];

    // per-wave LDS read bases
    const bf16* aRd0 = &As[(wm0 + l16) * LDS_STR];
    const bf16* bRd0 = &Bs[(wn0 + l16) * LDS_STR + (h << 3)];

    // ---- preload tile 0 into buffer 0 ----
    {
        uint4 a0 = *(const uint4*)Ap;
        uint4 a1 = *(const uint4*)(Ap + 8);
        uint4 b0 = *(const uint4*)Bp;
        uint4 b1 = *(const uint4*)(Bp + 8);
        *(uint4*)AsW       = a0;
        *(uint4*)(AsW + 8) = a1;
        *(uint4*)BsW       = b0;
        *(uint4*)(BsW + 8) = b1;
        Ap += 32; Bp += 32;
    }
    __syncthreads();

    auto compute_tile = [&](int cur) {
        const bf16* aB = aRd0 + cur * LDS_TILE;
        const bf16* bB = bRd0 + cur * LDS_TILE;
        v16bf afr[2], bfr[4];
#pragma unroll
        for (int mi = 0; mi < 2; ++mi) {
            union { v16bf v; unsigned u[8]; } f;
            const bf16* rp = aB + mi * 16 * LDS_STR;
#pragma unroll
            for (int r = 0; r < 8; ++r) {
                const int kk = ((r >> 2) << 4) + (h << 3) + ((r & 3) << 1);
                f.u[r] = *(const unsigned*)(rp + kk);
            }
            afr[mi] = f.v;
        }
#pragma unroll
        for (int ni = 0; ni < 4; ++ni) {
            union { v16bf v; unsigned u[8]; } f;
            const bf16* cp = bB + ni * 16 * LDS_STR;
#pragma unroll
            for (int r = 0; r < 8; ++r)
                f.u[r] = *(const unsigned*)(cp + ((r >> 2) << 4) + ((r & 3) << 1));
            bfr[ni] = f.v;
        }
#pragma unroll
        for (int mi = 0; mi < 2; ++mi)
#pragma unroll
            for (int ni = 0; ni < 4; ++ni)
                acc[mi][ni] = __builtin_amdgcn_wmma_f32_16x16x32_bf16(
                    false, afr[mi], false, bfr[ni], (short)0, acc[mi][ni], false, false);
    };

    const int nk = K >> 5;
    int cur = 0;
    for (int kt = 0; kt < nk - 1; ++kt) {
        // issue next tile's global loads early; they overlap the WMMA chain
        uint4 a0 = *(const uint4*)Ap;
        uint4 a1 = *(const uint4*)(Ap + 8);
        uint4 b0 = *(const uint4*)Bp;
        uint4 b1 = *(const uint4*)(Bp + 8);
        __builtin_prefetch(Ap + 96, 0, 0);   // ~3 tiles ahead, speculative
        __builtin_prefetch(Bp + 96, 0, 0);
        Ap += 32; Bp += 32;

        compute_tile(cur);

        const int nxt = cur ^ 1;
        bf16* aw = AsW + nxt * LDS_TILE;
        bf16* bw = BsW + nxt * LDS_TILE;
        *(uint4*)aw       = a0;
        *(uint4*)(aw + 8) = a1;
        *(uint4*)bw       = b0;
        *(uint4*)(bw + 8) = b1;
        __syncthreads();
        cur = nxt;
    }
    compute_tile(cur);   // last tile

    // ---- epilogue: bias + activation, dual-precision store ----
#pragma unroll
    for (int mi = 0; mi < 2; ++mi)
#pragma unroll
        for (int ni = 0; ni < 4; ++ni)
#pragma unroll
            for (int e = 0; e < 8; ++e) {
                const int r = bm + wm0 + mi * 16 + e + h * 8;
                const int c = bn + wn0 + ni * 16 + l16;
                if (c < N) {
                    float x = acc[mi][ni][e];
                    if (bias) x += bias[c];
                    if (act == 1)      x = fmaxf(x, 0.0f);
                    else if (act == 2) x = sigmoidf_(x);
                    const size_t o = (size_t)r * N + c;
                    if (Cf) Cf[o] = x;
                    if (Cb) Cb[o] = (bf16)x;
                }
            }
}

// ---------------------------------------------------------------------------
// Weight prep kernels
// ---------------------------------------------------------------------------
__global__ void gran_f32_to_bf16(const float* __restrict__ in, bf16* __restrict__ out, size_t n) {
    size_t i = (size_t)blockIdx.x * blockDim.x + threadIdx.x;
    if (i < n) out[i] = (bf16)in[i];
}

// in: f32 [kin, n] (row-major) -> out: bf16 [n, kout] (transposed, k zero-padded)
__global__ void gran_transpose_pad(const float* __restrict__ in, bf16* __restrict__ out,
                                   int kin, int kout, int n) {
    size_t i = (size_t)blockIdx.x * blockDim.x + threadIdx.x;
    size_t tot = (size_t)n * kout;
    if (i >= tot) return;
    int c = (int)(i / kout), k = (int)(i % kout);   // out[c][k]
    out[i] = (k < kin) ? (bf16)in[(size_t)k * n + c] : (bf16)0.0f;
}

// ---------------------------------------------------------------------------
// Elementwise / data-movement kernels
// ---------------------------------------------------------------------------
// X[4B,256] bf16: gathered embeddings; Event rows add relu-text projection
__global__ void gran_build_x(const int* __restrict__ a_ids, const int* __restrict__ ev_ids,
                             const int* __restrict__ b_ids, const int* __restrict__ c_ids,
                             const float* __restrict__ ent, const float* __restrict__ txt,
                             bf16* __restrict__ X, int Bn) {
    size_t i = (size_t)blockIdx.x * blockDim.x + threadIdx.x;
    size_t tot = (size_t)4 * Bn * 256;
    if (i >= tot) return;
    int r = (int)(i >> 8), c = (int)(i & 255);
    int g = r / Bn, j = r % Bn;
    int id = (g == 0) ? a_ids[j] : (g == 1) ? ev_ids[j] : (g == 2) ? b_ids[j] : c_ids[j];
    float v = ent[(size_t)id * 256 + c];
    if (g == 1) v += txt[(size_t)j * 256 + c];
    X[i] = (bf16)v;
}

// ein[3B,544] bf16: state[src]-state[dst] (512) | one-hot edge type (8) | zero pad
__global__ void gran_build_ein(const float* __restrict__ state, bf16* __restrict__ ein, int Bn) {
    size_t i = (size_t)blockIdx.x * blockDim.x + threadIdx.x;
    size_t tot = (size_t)3 * Bn * 544;
    if (i >= tot) return;
    int e = (int)(i / 544), c = (int)(i % 544);
    int g = e / Bn, j = e % Bn;
    int src = (g == 0) ? j : (Bn + j);
    int dst = (g + 1) * Bn + j;
    float v;
    if (c < 512)      v = state[(size_t)src * 512 + c] - state[(size_t)dst * 512 + c];
    else if (c < 520) v = ((c - 512) == g) ? 1.0f : 0.0f;
    else              v = 0.0f;
    ein[i] = (bf16)v;
}

// agg[4B,512] bf16: scatter (each dst has exactly one incoming edge); A-rows get 0
__global__ void gran_agg(const float* __restrict__ msg, const float* __restrict__ att,
                         bf16* __restrict__ agg, int Bn) {
    size_t i = (size_t)blockIdx.x * blockDim.x + threadIdx.x;
    size_t tot = (size_t)4 * Bn * 512;
    if (i >= tot) return;
    int r = (int)(i >> 9), c = (int)(i & 511);
    if (r < Bn) { agg[i] = (bf16)0.0f; return; }
    size_t e = (size_t)(r - Bn) * 512 + c;
    agg[i] = (bf16)(msg[e] * att[e]);
}

// torch GRUCell update; writes state in f32 and bf16
__global__ void gran_gru(const float* __restrict__ gi, const float* __restrict__ gh,
                         float* __restrict__ state, bf16* __restrict__ state_b, int Bn) {
    size_t i = (size_t)blockIdx.x * blockDim.x + threadIdx.x;
    size_t tot = (size_t)4 * Bn * 512;
    if (i >= tot) return;
    int r = (int)(i >> 9), c = (int)(i & 511);
    size_t g = (size_t)r * 1536 + c;
    float ir = gi[g], iz = gi[g + 512], in_ = gi[g + 1024];
    float hr = gh[g], hz = gh[g + 512], hn  = gh[g + 1024];
    float rg = sigmoidf_(ir + hr);
    float zg = sigmoidf_(iz + hz);
    float ng = tanhf(in_ + rg * hn);
    float s  = state[i];
    float ns = (1.0f - zg) * ng + zg * s;
    state[i] = ns;
    state_b[i] = (bf16)ns;
}

__global__ void gran_relu_state(float* __restrict__ state, bf16* __restrict__ state_b, size_t n) {
    size_t i = (size_t)blockIdx.x * blockDim.x + threadIdx.x;
    if (i >= n) return;
    float v = fmaxf(state[i], 0.0f);
    state[i] = v;
    state_b[i] = (bf16)v;
}

// diff[B,512] bf16 = state[A rows] - state[B rows]
__global__ void gran_diff(const float* __restrict__ state, bf16* __restrict__ dif, int Bn) {
    size_t i = (size_t)blockIdx.x * blockDim.x + threadIdx.x;
    size_t tot = (size_t)Bn * 512;
    if (i >= tot) return;
    dif[i] = (bf16)(state[i] - state[i + (size_t)2 * Bn * 512]);
}

// ---------------------------------------------------------------------------
extern "C" void kernel_launch(void* const* d_in, const int* in_sizes, int n_in,
                              void* d_out, int out_size, void* d_ws, size_t ws_size,
                              hipStream_t stream)
{
    (void)in_sizes; (void)n_in; (void)out_size; (void)ws_size;

    const int Bn = 16384, D = 256, H = 512, T = 768, L = 2, R = 32;
    const int KE = 544;                 // H + EF(8) padded to multiple of 32
    const int N4 = 4 * Bn, N3 = 3 * Bn;

    const int*   a_ids   = (const int*)d_in[0];
    const int*   ev_ids  = (const int*)d_in[1];
    const int*   b_ids   = (const int*)d_in[2];
    const int*   c_ids   = (const int*)d_in[3];
    const float* text_ab = (const float*)d_in[4];
    const float* ent_emb = (const float*)d_in[5];
    const float* text_W  = (const float*)d_in[6];
    const float* text_b  = (const float*)d_in[7];
    const float* proj_W  = (const float*)d_in[8];
    const float* proj_b  = (const float*)d_in[9];
    const float* msg_W1  = (const float*)d_in[10];
    const float* msg_b1  = (const float*)d_in[11];
    const float* msg_W2  = (const float*)d_in[12];
    const float* msg_b2  = (const float*)d_in[13];
    const float* att_W1  = (const float*)d_in[14];
    const float* att_b1  = (const float*)d_in[15];
    const float* att_W2  = (const float*)d_in[16];
    const float* att_b2  = (const float*)d_in[17];
    const float* gru_Wih = (const float*)d_in[18];
    const float* gru_Whh = (const float*)d_in[19];
    const float* gru_bih = (const float*)d_in[20];
    const float* gru_bhh = (const float*)d_in[21];
    const float* head_W1 = (const float*)d_in[22];
    const float* head_b1 = (const float*)d_in[23];
    const float* head_W2 = (const float*)d_in[24];
    const float* head_b2 = (const float*)d_in[25];

    char* wp = (char*)d_ws;
    auto alloc = [&](size_t bytes) -> void* {
        void* p = (void*)wp;
        wp += (bytes + 255) & ~size_t(255);
        return p;
    };

    // all weight buffers hold B^T (row-major [N, Kpadded]) in bf16
    bf16*  wTextA = (bf16*)alloc((size_t)Bn * T * 2);
    bf16*  wTextW = (bf16*)alloc((size_t)D * T * 2);            // [256, 768]
    bf16*  wProjW = (bf16*)alloc((size_t)H * D * 2);            // [512, 256]
    bf16*  wMsg1  = (bf16*)alloc((size_t)L * H * KE * 2);       // [512, 544] x L
    bf16*  wMsg2  = (bf16*)alloc((size_t)L * H * H * 2);        // [512, 512] x L
    bf16*  wAtt1  = (bf16*)alloc((size_t)L * (H / 2) * KE * 2); // [256, 544] x L
    bf16*  wAtt2  = (bf16*)alloc((size_t)L * H * (H / 2) * 2);  // [512, 256] x L
    bf16*  wGih   = (bf16*)alloc((size_t)L * 3 * H * H * 2);    // raw [1536, 512] == (Wih.T)^T
    bf16*  wGhh   = (bf16*)alloc((size_t)L * 3 * H * H * 2);
    bf16*  wHead1 = (bf16*)alloc((size_t)H * H * 2);            // [512, 512]
    bf16*  wHead2 = (bf16*)alloc((size_t)R * H * 2);            // [32, 512]
    float* txt    = (float*)alloc((size_t)Bn * D * 4);
    bf16*  X      = (bf16*)alloc((size_t)N4 * D * 2);
    float* stF    = (float*)alloc((size_t)N4 * H * 4);
    bf16*  stB    = (bf16*)alloc((size_t)N4 * H * 2);
    bf16*  ein    = (bf16*)alloc((size_t)N3 * KE * 2);
    bf16*  hid    = (bf16*)alloc((size_t)N3 * H * 2);
    float* msg    = (float*)alloc((size_t)N3 * H * 4);
    float* att    = (float*)alloc((size_t)N3 * H * 4);
    bf16*  agg    = (bf16*)alloc((size_t)N4 * H * 2);
    float* gi     = (float*)alloc((size_t)N4 * 3 * H * 4);
    float* gh     = (float*)alloc((size_t)N4 * 3 * H * 4);
    bf16*  dif    = (bf16*)alloc((size_t)Bn * H * 2);
    bf16*  hidh   = (bf16*)alloc((size_t)Bn * H * 2);

    auto nb = [](size_t n) { return (unsigned)((n + 255) / 256); };
    auto conv = [&](const float* in, bf16* out, size_t n) {
        gran_f32_to_bf16<<<nb(n), 256, 0, stream>>>(in, out, n);
    };
    auto tpad = [&](const float* in, bf16* out, int kin, int kout, int n) {
        gran_transpose_pad<<<nb((size_t)n * kout), 256, 0, stream>>>(in, out, kin, kout, n);
    };
    auto gemm = [&](const bf16* A, const bf16* Bt, const float* bias,
                    float* Cf, bf16* Cb, int M, int N, int K, int act) {
        dim3 g((N + 127) / 128, (M + 127) / 128);
        gran_gemm_bf16<<<g, 256, 0, stream>>>(A, Bt, bias, Cf, Cb, M, N, K, act);
    };

    // ---- weight prep: bf16, transposed to [N,K], K padded to mult of 32 ----
    conv(text_ab, wTextA, (size_t)Bn * T);
    tpad(text_W,  wTextW, T, T, D);                           // [768,256] -> [256,768]
    tpad(proj_W,  wProjW, D, D, H);                           // [256,512] -> [512,256]
    conv(gru_Wih, wGih, (size_t)L * 3 * H * H);               // already [N,K]
    conv(gru_Whh, wGhh, (size_t)L * 3 * H * H);
    tpad(head_W1, wHead1, H, H, H);                           // [512,512] -> [512,512]
    tpad(head_W2, wHead2, H, H, R);                           // [512,32]  -> [32,512]
    for (int ii = 0; ii < L; ++ii) {
        tpad(msg_W1 + (size_t)ii * (H + 8) * H,       wMsg1 + (size_t)ii * H * KE,       H + 8, KE, H);
        tpad(att_W1 + (size_t)ii * (H + 8) * (H / 2), wAtt1 + (size_t)ii * (H / 2) * KE, H + 8, KE, H / 2);
        tpad(msg_W2 + (size_t)ii * H * H,             wMsg2 + (size_t)ii * H * H,        H, H, H);
        tpad(att_W2 + (size_t)ii * (H / 2) * H,       wAtt2 + (size_t)ii * H * (H / 2),  H / 2, H / 2, H);
    }

    // ---- text fuse + gather + input projection ----
    gemm(wTextA, wTextW, text_b, txt, nullptr, Bn, D, T, 1);                 // relu(text@W+b)
    gran_build_x<<<nb((size_t)N4 * D), 256, 0, stream>>>(
        a_ids, ev_ids, b_ids, c_ids, ent_emb, txt, X, Bn);
    gemm(X, wProjW, proj_b, stF, stB, N4, H, D, 1);                          // state = relu(proj)

    // ---- L layers x NUM_PROP propagation steps ----
    for (int ii = 0; ii < L; ++ii) {
        if (ii > 0)
            gran_relu_state<<<nb((size_t)N4 * H), 256, 0, stream>>>(stF, stB, (size_t)N4 * H);
        for (int p = 0; p < 2; ++p) {
            gran_build_ein<<<nb((size_t)N3 * KE), 256, 0, stream>>>(stF, ein, Bn);
            gemm(ein, wMsg1 + (size_t)ii * H * KE, msg_b1 + ii * H, nullptr, hid, N3, H, KE, 1);
            gemm(hid, wMsg2 + (size_t)ii * H * H, msg_b2 + ii * H, msg, nullptr, N3, H, H, 0);
            gemm(ein, wAtt1 + (size_t)ii * (H / 2) * KE, att_b1 + ii * (H / 2), nullptr, hid, N3, H / 2, KE, 1);
            gemm(hid, wAtt2 + (size_t)ii * H * (H / 2), att_b2 + ii * H, att, nullptr, N3, H, H / 2, 2);
            gran_agg<<<nb((size_t)N4 * H), 256, 0, stream>>>(msg, att, agg, Bn);
            gemm(agg, wGih + (size_t)ii * 3 * H * H, gru_bih + ii * 3 * H, gi, nullptr, N4, 3 * H, H, 0);
            gemm(stB, wGhh + (size_t)ii * 3 * H * H, gru_bhh + ii * 3 * H, gh, nullptr, N4, 3 * H, H, 0);
            gran_gru<<<nb((size_t)N4 * H), 256, 0, stream>>>(gi, gh, stF, stB, Bn);
        }
    }

    // ---- relation head ----
    gran_diff<<<nb((size_t)Bn * H), 256, 0, stream>>>(stF, dif, Bn);
    gemm(dif, wHead1, head_b1, nullptr, hidh, Bn, H, H, 1);
    gemm(hidh, wHead2, head_b2, (float*)d_out, nullptr, Bn, R, H, 0);
}
